// CodebookDecoder_1632087573351
// MI455X (gfx1250) — compile-verified, matrix operand order checked by reading, not run on hardware
//
#include <hip/hip_runtime.h>

// Problem dimensions (fixed by the reference).
#define D_DIM   256
#define N_ENT   50000    // 3125 groups of 16
#define K_CODES 1024
#define Q_ROWS  4096
#define GATHER_BLOCKS 512

typedef __attribute__((ext_vector_type(16))) __bf16 v16bf;
typedef __attribute__((ext_vector_type(4)))  __bf16 v4bf;
typedef __attribute__((ext_vector_type(8)))  float  v8f;
typedef __attribute__((ext_vector_type(2)))  float  v2f;

union FragBF {
    v16bf v;
    uint4 q[2];
};

// ---- fragment loaders -------------------------------------------------------
// bf16 16x32 A-layout (and B for C = A*B^T, both row-major):
// lane l: row = l&15 ; K-chunk base ks = (l>>4)*8 ; holds [ks..ks+7] and [ks+16..ks+23]
__device__ inline v16bf load_frag_bf16(const __bf16* __restrict__ base,
                                       int rowBase, int kBase, int lane) {
    int r  = lane & 15;
    int ks = (lane >> 4) << 3;                 // 0 or 8
    const __bf16* p = base + (size_t)(rowBase + r) * D_DIM + kBase + ks;
    FragBF f;
    f.q[0] = *(const uint4*)(p);               // 8 bf16: K = kBase+ks .. +7
    f.q[1] = *(const uint4*)(p + 16);          // 8 bf16: K = kBase+ks+16 .. +23
    return f.v;
}

// f32 16x4 A-layout (and B for C = A*B^T): lane l: row = l&15, K = kBase + 2*(l>>4) + {0,1}
__device__ inline v2f load_frag_f32(const float* __restrict__ base,
                                    int rowBase, int kBase, int lane) {
    int r  = lane & 15;
    int ks = (lane >> 4) << 1;                 // 0 or 2
    return *(const v2f*)(base + (size_t)(rowBase + r) * D_DIM + kBase + ks);
}

// ---- f32 -> bf16 conversion (vectorized by 4) -------------------------------
__global__ void cvt_bf16_kernel(const float* __restrict__ in,
                                __bf16* __restrict__ out, int n4) {
    int i = blockIdx.x * blockDim.x + threadIdx.x;
    if (i < n4) {
        float4 f = ((const float4*)in)[i];
        v4bf o;
        o[0] = (__bf16)f.x; o[1] = (__bf16)f.y;
        o[2] = (__bf16)f.z; o[3] = (__bf16)f.w;
        ((v4bf*)out)[i] = o;
    }
}

// ---- codebook squared norms -------------------------------------------------
__global__ void c2_kernel(const float* __restrict__ cb, float* __restrict__ c2) {
    int k = blockIdx.x * blockDim.x + threadIdx.x;
    if (k < K_CODES) {
        const float* row = cb + (size_t)k * D_DIM;
        float s = 0.f;
        for (int d = 0; d < D_DIM; d += 4) {
            float4 v = *(const float4*)(row + d);
            s += v.x * v.x + v.y * v.y + v.z * v.z + v.w * v.w;
        }
        c2[k] = s;
    }
}

// ---- nearest-code search: f32 WMMA (precision-critical, small GEMM) ---------
// One wave handles 16 entities vs all 1024 codes; 4 code-tiles (64 codes) per strip.
__global__ void __launch_bounds__(256) argmin_kernel(
        const float* __restrict__ ent, const float* __restrict__ cb,
        const float* __restrict__ c2, int* __restrict__ idx_out,
        float* __restrict__ idx_f_out) {
    int wave = threadIdx.x >> 5;
    int lane = threadIdx.x & 31;
    int grp  = blockIdx.x * 8 + wave;          // 16-entity group
    if (grp >= N_ENT / 16) return;             // uniform per-wave: EXEC stays all-1s
    int rowBase = grp * 16;

    float best[8];
    int   bidx[8];
#pragma unroll
    for (int j = 0; j < 8; j++) { best[j] = 3.4e38f; bidx[j] = 0; }

    for (int cs = 0; cs < K_CODES / 64; cs++) {      // 16 strips of 64 codes
        v8f acc[4];
#pragma unroll
        for (int t = 0; t < 4; t++) acc[t] = {};
#pragma unroll 4
        for (int k = 0; k < D_DIM / 4; k++) {        // 64 K-steps of 4
            v2f a = load_frag_f32(ent, rowBase, k * 4, lane);
#pragma unroll
            for (int t = 0; t < 4; t++) {
                v2f b = load_frag_f32(cb, cs * 64 + t * 16, k * 4, lane);
                acc[t] = __builtin_amdgcn_wmma_f32_16x16x4_f32(
                    false, a, false, b, (short)0, acc[t], false, false);
            }
        }
#pragma unroll
        for (int t = 0; t < 4; t++) {
            int   code = cs * 64 + t * 16 + (lane & 15);
            float cc   = c2[code];
#pragma unroll
            for (int j = 0; j < 8; j++) {
                float d2 = cc - 2.0f * acc[t][j];    // x2 per-row constant: drop it
                if (d2 < best[j] || (d2 == best[j] && code < bidx[j])) {
                    best[j] = d2; bidx[j] = code;
                }
            }
        }
    }
    // reduce across the 16 lanes of each half (C layout: lane=col, vgpr j=row)
#pragma unroll
    for (int off = 8; off >= 1; off >>= 1) {
#pragma unroll
        for (int j = 0; j < 8; j++) {
            float ov = __shfl_xor(best[j], off, 32);
            int   oi = __shfl_xor(bidx[j], off, 32);
            if (ov < best[j] || (ov == best[j] && oi < bidx[j])) {
                best[j] = ov; bidx[j] = oi;
            }
        }
    }
    if ((lane & 15) == 0) {
        int rbase = rowBase + ((lane >> 4) << 3);    // rows 0-7 (lane0) / 8-15 (lane16)
#pragma unroll
        for (int j = 0; j < 8; j++) {
            idx_out[rbase + j]   = bidx[j];
            idx_f_out[rbase + j] = (float)bidx[j];
        }
    }
}

// ---- gather quantized rows (bf16) + partial VQ-loss sums --------------------
__global__ void __launch_bounds__(256) gather_loss_kernel(
        const float* __restrict__ ent, const float* __restrict__ cb,
        const int* __restrict__ idx, __bf16* __restrict__ quantB,
        float* __restrict__ partial) {
    float s = 0.f;
    size_t total  = (size_t)N_ENT * D_DIM;
    size_t stride = (size_t)gridDim.x * blockDim.x;
    for (size_t e = (size_t)blockIdx.x * blockDim.x + threadIdx.x; e < total; e += stride) {
        size_t n = e >> 8;                     // / 256
        int    d = (int)(e & 255);
        float qv = cb[(size_t)idx[n] * D_DIM + d];
        float xv = ent[e];
        float df = qv - xv;
        s += df * df;
        quantB[e] = (__bf16)qv;
    }
    __shared__ float sm[256];
    sm[threadIdx.x] = s;
    __syncthreads();
    for (int st = 128; st > 0; st >>= 1) {
        if ((int)threadIdx.x < st) sm[threadIdx.x] += sm[threadIdx.x + st];
        __syncthreads();
    }
    if (threadIdx.x == 0) partial[blockIdx.x] = sm[0];
}

__global__ void finalize_loss_kernel(const float* __restrict__ partial,
                                     float* __restrict__ out_loss) {
    __shared__ float sm[GATHER_BLOCKS];
    sm[threadIdx.x] = partial[threadIdx.x];
    __syncthreads();
    for (int st = GATHER_BLOCKS / 2; st > 0; st >>= 1) {
        if ((int)threadIdx.x < st) sm[threadIdx.x] += sm[threadIdx.x + st];
        __syncthreads();
    }
    if (threadIdx.x == 0)
        out_loss[0] = 1.25f * sm[0] / (float)((size_t)N_ENT * D_DIM);
}

// ---- score = query @ quantized^T : bf16 WMMA, output-write bound ------------
// Wave tile: 64 (M, query rows) x 16 (N, entities). 8 waves share one 64-row
// query tile (WGP$-resident); quantized (25.6MB) + query (2MB) stay in L2.
__global__ void __launch_bounds__(256) score_kernel(
        const __bf16* __restrict__ qB, const __bf16* __restrict__ quantB,
        float* __restrict__ out) {
    int wave  = threadIdx.x >> 5;
    int lane  = threadIdx.x & 31;
    int mBase = blockIdx.x * 64;
    int strip = blockIdx.y * 8 + wave;         // 16-entity output strip
    if (strip >= N_ENT / 16) return;           // uniform per-wave
    int nBase = strip * 16;

    v8f acc[4];
#pragma unroll
    for (int mi = 0; mi < 4; mi++) acc[mi] = {};

#pragma unroll
    for (int kk = 0; kk < D_DIM / 32; kk++) {  // 8 K-steps of 32
        v16bf b = load_frag_bf16(quantB, nBase, kk * 32, lane);
#pragma unroll
        for (int mi = 0; mi < 4; mi++) {
            v16bf a = load_frag_bf16(qB, mBase + mi * 16, kk * 32, lane);
            acc[mi] = __builtin_amdgcn_wmma_f32_16x16x32_bf16(
                false, a, false, b, (short)0, acc[mi], false, false);
        }
    }
    int col  = nBase + (lane & 15);
    int rOff = (lane >> 4) << 3;
#pragma unroll
    for (int mi = 0; mi < 4; mi++) {
#pragma unroll
        for (int j = 0; j < 8; j++) {
            int row = mBase + mi * 16 + rOff + j;
            out[(size_t)row * N_ENT + col] = acc[mi][j];
        }
    }
}

// ---- host-side orchestration ------------------------------------------------
extern "C" void kernel_launch(void* const* d_in, const int* in_sizes, int n_in,
                              void* d_out, int out_size, void* d_ws, size_t ws_size,
                              hipStream_t stream) {
    (void)in_sizes; (void)n_in; (void)out_size; (void)ws_size;
    const float* query    = (const float*)d_in[0];   // [4096, 256]
    const float* entity   = (const float*)d_in[1];   // [50000, 256]
    const float* codebook = (const float*)d_in[2];   // [1024, 256]
    float* out = (float*)d_out;

    // workspace carve-out (≈28 MB)
    char*  ws  = (char*)d_ws;
    size_t off = 0;
    auto take = [&](size_t bytes) {
        char* p = ws + off;
        off += (bytes + 255) & ~(size_t)255;
        return p;
    };
    __bf16* qB      = (__bf16*)take((size_t)Q_ROWS * D_DIM * sizeof(__bf16));
    __bf16* quantB  = (__bf16*)take((size_t)N_ENT * D_DIM * sizeof(__bf16));
    float*  c2      = (float*)take(K_CODES * sizeof(float));
    int*    idx     = (int*)take((size_t)N_ENT * sizeof(int));
    float*  partial = (float*)take(GATHER_BLOCKS * sizeof(float));

    // output layout: score [Q*N], vq_loss [1], nearest_idx [N] (as float)
    float* out_score = out;
    float* out_loss  = out + (size_t)Q_ROWS * N_ENT;
    float* out_idx   = out_loss + 1;

    // 1) query f32 -> bf16
    {
        int n4 = Q_ROWS * D_DIM / 4;
        cvt_bf16_kernel<<<(n4 + 255) / 256, 256, 0, stream>>>(query, qB, n4);
    }
    // 2) codebook squared norms
    c2_kernel<<<K_CODES / 256, 256, 0, stream>>>(codebook, c2);
    // 3) nearest-code argmin (f32 WMMA), writes int idx (ws) + float idx (d_out)
    {
        int groups = N_ENT / 16;               // 3125
        argmin_kernel<<<(groups + 7) / 8, 256, 0, stream>>>(
            entity, codebook, c2, idx, out_idx);
    }
    // 4) gather quantized (bf16) + partial loss sums
    gather_loss_kernel<<<GATHER_BLOCKS, 256, 0, stream>>>(
        entity, codebook, idx, quantB, partial);
    // 5) deterministic loss reduction
    finalize_loss_kernel<<<1, GATHER_BLOCKS, 0, stream>>>(partial, out_loss);
    // 6) big score GEMM (bf16 WMMA)
    {
        dim3 grid(Q_ROWS / 64, (N_ENT / 16 + 7) / 8);
        score_kernel<<<grid, 256, 0, stream>>>(qB, quantB, out_score);
    }
}
